// EncoderRNN_31868657336601
// MI455X (gfx1250) — matrix operand order repaired and checked by reading, not compiled, not run on hardware
//
#include <hip/hip_runtime.h>
#include <hip/hip_bf16.h>

typedef __bf16 bf16;
typedef __attribute__((ext_vector_type(16))) __bf16 v16bf;
typedef __attribute__((ext_vector_type(8)))  __bf16 v8bf;
typedef __attribute__((ext_vector_type(8)))  float  v8f;
typedef __attribute__((ext_vector_type(4)))  int    v4i;

// Address-space-qualified v4i pointers for the async global->LDS builtin
// (signature leaked by clang diagnostic: (v4i AS1*, v4i AS3*, imm, imm)).
typedef __attribute__((address_space(1))) v4i as1_v4i;
typedef __attribute__((address_space(3))) v4i as3_v4i;

constexpr int T_ = 256, B_ = 128, I_ = 512, H_ = 1024, L_ = 4, G_ = 3 * H_;

#if __has_builtin(__builtin_amdgcn_global_load_async_to_lds_b128)
#define HAVE_ASYNC_LDS 1
#else
#define HAVE_ASYNC_LDS 0
#endif

// ---------------------------------------------------------------------------
// Fragment loaders (CDNA5 wave32 WMMA VGPR layouts, CDNA5 ISA)
// A: 16x32 bf16, lane l holds row (l&15); K-chunks [ (l>>4)*8 .. +7 ] and
//    [ 16+(l>>4)*8 .. +7 ]  -> two 16B loads.
// B: 32x16 bf16, lane l holds col (l&15); K-chunks [ (l>>4)*8 .. +7 ] and
//    [ 16+(l>>4)*8 .. +7 ]  -> two 16B loads (same order as A).
// ---------------------------------------------------------------------------
__device__ __forceinline__ v16bf pack_frag(v8bf lo, v8bf hv) {
  v16bf a;
#pragma unroll
  for (int i = 0; i < 8; ++i) { a[i] = lo[i]; a[i + 8] = hv[i]; }
  return a;
}

__device__ __forceinline__ v16bf load_a_frag(const bf16* A, int lda, int row0,
                                             int k0, int lane) {
  const int r  = row0 + (lane & 15);
  const int hi = lane >> 4;
  const bf16* p = A + (size_t)r * lda + k0 + hi * 8;
  return pack_frag(*(const v8bf*)p, *(const v8bf*)(p + 16));
}

// A fragment sourced from an LDS tile laid out [rows][H_]
__device__ __forceinline__ v16bf load_a_frag_lds(const bf16* S, int row0,
                                                 int k0, int lane) {
  const int r  = row0 + (lane & 15);
  const int hi = lane >> 4;
  const bf16* p = S + r * H_ + k0 + hi * 8;
  return pack_frag(*(const v8bf*)p, *(const v8bf*)(p + 16));
}

__device__ __forceinline__ v16bf load_b_frag(const bf16* W, int ldb, int col0,
                                             int k0, int lane) {
  const int c  = col0 + (lane & 15);
  const int kh = lane >> 4;
  const bf16* p = W + (size_t)c * ldb + k0 + kh * 8;
  return pack_frag(*(const v8bf*)p, *(const v8bf*)(p + 16));
}

__device__ __forceinline__ v8f wmma_bf16(v16bf a, v16bf b, v8f c) {
  return __builtin_amdgcn_wmma_f32_16x16x32_bf16(
      /*neg_a=*/false, a, /*neg_b=*/false, b,
      /*c_mod=*/(short)0, c, /*reuse_a=*/false, /*reuse_b=*/false);
}

__device__ __forceinline__ float sigmoid_(float x) {
  return 1.0f / (1.0f + __expf(-x));
}
__device__ __forceinline__ float tanh_(float x) {
  float e = __expf(2.0f * x);
  return (e - 1.0f) / (e + 1.0f);
}

// ---------------------------------------------------------------------------
// Elementwise converts / init
// ---------------------------------------------------------------------------
__global__ void f32_to_bf16(const float* __restrict__ in, bf16* __restrict__ out,
                            long long n) {
  long long i = (long long)blockIdx.x * blockDim.x + threadIdx.x;
  if (i < n) out[i] = (bf16)in[i];
}

__global__ void init_h(const float* __restrict__ h0, float* __restrict__ hstate,
                       bf16* __restrict__ hbf, int n) {
  int i = blockIdx.x * blockDim.x + threadIdx.x;
  if (i < n) {
    float v = h0[i];
    hstate[i] = v;
    hbf[i] = (bf16)v;
  }
}

// ---------------------------------------------------------------------------
// gi GEMM: out[m, g] = sum_k A[m,k] * W[g,k] + bias[g]   (bf16 in, bf16 out,
// f32 accumulate).  Block tile 64x64, 8 waves (2 M x 4 N), wave = 32x16.
// ---------------------------------------------------------------------------
__global__ __launch_bounds__(256) void gi_gemm(const bf16* __restrict__ A, int lda,
                                               const bf16* __restrict__ W,
                                               const float* __restrict__ bias,
                                               bf16* __restrict__ out) {
  const int lane = threadIdx.x & 31;
  const int wid  = threadIdx.x >> 5;
  const int wm = wid & 1, wn = wid >> 1;
  const int m0 = blockIdx.x * 64 + wm * 32;
  const int n0 = blockIdx.y * 64 + wn * 16;

  v8f c0 = {}, c1 = {};
  for (int k = 0; k < lda; k += 32) {
    v16bf b  = load_b_frag(W, lda, n0, k, lane);
    v16bf a0 = load_a_frag(A, lda, m0, k, lane);
    v16bf a1 = load_a_frag(A, lda, m0 + 16, k, lane);
    c0 = wmma_bf16(a0, b, c0);
    c1 = wmma_bf16(a1, b, c1);
  }

  const int col   = n0 + (lane & 15);
  const float bv  = bias[col];
  const int rbase = m0 + 8 * (lane >> 4);
#pragma unroll
  for (int v = 0; v < 8; ++v) {
    out[(size_t)(rbase + v) * G_ + col]      = (bf16)(c0[v] + bv);
    out[(size_t)(rbase + 16 + v) * G_ + col] = (bf16)(c1[v] + bv);
  }
}

// ---------------------------------------------------------------------------
// Fused recurrent step: stage the block's 32x1024 h tile into LDS with the
// CDNA5 async global->LDS path, then gh = h @ w_hh.T (3 gate tiles, K=1024,
// WMMA; A from LDS via ds_load, B streamed from L2), then full GRU gating.
// Block tile: 32 batch rows x 64 hidden cols; 8 waves (2 M x 4 N),
// wave = 16x16 per gate (3 accumulators).
// ---------------------------------------------------------------------------
__global__ __launch_bounds__(256) void gru_step(
    const bf16* __restrict__ h_in, bf16* __restrict__ h_out,
    const bf16* __restrict__ whh,   // [3H, H] bf16, this layer
    const bf16* __restrict__ gi_t,  // [B, 3H] bf16, this timestep (bi folded in)
    const float* __restrict__ bhh,  // [3H] f32, this layer
    const int* __restrict__ lens,   // [B]
    float* __restrict__ hstate,     // [B, H] f32 running state (in-place ok)
    float* __restrict__ y_f32,      // last layer: d_out ys base, else null
    bf16* __restrict__ y_bf16,      // other layers: next-layer input, else null
    int t, int last) {
  __shared__ bf16 hs[32 * H_];  // 64 KB: this block's 32 rows of h (bf16)

  const int lane = threadIdx.x & 31;
  const int wid  = threadIdx.x >> 5;
  const int wm = wid & 1, wn = wid >> 1;
  const int m0 = wm * 16;                    // batch-row offset inside LDS tile
  const int n0 = blockIdx.y * 64 + wn * 16;  // hidden cols
  const int rowblk = blockIdx.x * 32;        // global batch-row base

  // ---- stage h tile: 32 rows x 1024 cols = 4096 x 16B chunks, 16/thread ----
  {
    const bf16* src = h_in + (size_t)rowblk * H_;
#pragma unroll
    for (int i = 0; i < 16; ++i) {
      const int elem = (threadIdx.x + i * 256) * 8;
#if HAVE_ASYNC_LDS
      __builtin_amdgcn_global_load_async_to_lds_b128(
          (as1_v4i*)(src + elem), (as3_v4i*)(hs + elem), 0, 0);
#else
      *(v8bf*)(hs + elem) = *(const v8bf*)(src + elem);
#endif
    }
#if HAVE_ASYNC_LDS
#if __has_builtin(__builtin_amdgcn_s_wait_asynccnt)
    __builtin_amdgcn_s_wait_asynccnt(0);
#else
    asm volatile("s_wait_asynccnt 0x0" ::: "memory");
#endif
#endif
    __syncthreads();
  }

  // ---- K loop: A from LDS, B (3 gates) from global/L2 ----
  v8f cr = {}, cz = {}, cn = {};
  for (int k = 0; k < H_; k += 32) {
    v16bf a  = load_a_frag_lds(hs, m0, k, lane);
    v16bf br = load_b_frag(whh, H_, n0,          k, lane);
    v16bf bz = load_b_frag(whh, H_, n0 + H_,     k, lane);
    v16bf bn = load_b_frag(whh, H_, n0 + 2 * H_, k, lane);
    cr = wmma_bf16(a, br, cr);
    cz = wmma_bf16(a, bz, cz);
    cn = wmma_bf16(a, bn, cn);
  }

  // ---- gating epilogue ----
  const int col   = n0 + (lane & 15);
  const float bhr = bhh[col];
  const float bhz = bhh[H_ + col];
  const float bhn = bhh[2 * H_ + col];
  const int rbase = rowblk + m0 + 8 * (lane >> 4);

#pragma unroll
  for (int v = 0; v < 8; ++v) {
    const int row = rbase + v;
    const bf16* g = gi_t + (size_t)row * G_;
    const float r = sigmoid_((float)g[col]        + cr[v] + bhr);
    const float z = sigmoid_((float)g[H_ + col]   + cz[v] + bhz);
    const float n = tanh_((float)g[2 * H_ + col]  + r * (cn[v] + bhn));
    const size_t hix = (size_t)row * H_ + col;
    const float hp = hstate[hix];
    const float hn = (1.0f - z) * n + z * hp;
    const bool  msk = t < lens[row];
    const float hnext = msk ? hn : hp;
    const float y     = msk ? hn : 0.0f;
    hstate[hix] = hnext;
    h_out[hix]  = (bf16)hnext;
    const size_t yix = (size_t)t * (B_ * H_) + hix;
    if (last) y_f32[yix] = y;
    else      y_bf16[yix] = (bf16)y;
  }
}

// ---------------------------------------------------------------------------
extern "C" void kernel_launch(void* const* d_in, const int* in_sizes, int n_in,
                              void* d_out, int out_size, void* d_ws, size_t ws_size,
                              hipStream_t stream) {
  (void)in_sizes; (void)n_in; (void)out_size; (void)ws_size;

  const float* x    = (const float*)d_in[0];  // (T,B,I)
  const float* hid  = (const float*)d_in[1];  // (L,B,H)
  const float* wih0 = (const float*)d_in[2];  // (3H,I)
  const float* wihr = (const float*)d_in[3];  // (L-1,3H,H)
  const float* whh  = (const float*)d_in[4];  // (L,3H,H)
  const float* bih  = (const float*)d_in[5];  // (L,3H)
  const float* bhh  = (const float*)d_in[6];  // (L,3H)
  const int*   lens = (const int*)d_in[8];    // (B,)
  float* out = (float*)d_out;                 // ys (T,B,H) ++ h_finals (L,B,H)

  // Workspace carve-up (256B aligned)
  char* ws = (char*)d_ws;
  auto alloc = [&](size_t bytes) -> char* {
    char* p = ws;
    ws += (bytes + 255) & ~(size_t)255;
    return p;
  };
  bf16* WIH0 = (bf16*)alloc((size_t)G_ * I_ * 2);
  bf16* WIHR = (bf16*)alloc((size_t)(L_ - 1) * G_ * H_ * 2);
  bf16* WHH  = (bf16*)alloc((size_t)L_ * G_ * H_ * 2);
  bf16* XBUF = (bf16*)alloc((size_t)T_ * B_ * H_ * 2);  // layer in/out (bf16)
  bf16* GI   = (bf16*)alloc((size_t)T_ * B_ * G_ * 2);  // gate pre-acts
  float* HST = (float*)alloc((size_t)B_ * H_ * 4);
  bf16* HBF0 = (bf16*)alloc((size_t)B_ * H_ * 2);
  bf16* HBF1 = (bf16*)alloc((size_t)B_ * H_ * 2);

  auto cvt = [&](const float* src, bf16* dst, long long n) {
    int blocks = (int)((n + 255) / 256);
    f32_to_bf16<<<blocks, 256, 0, stream>>>(src, dst, n);
  };

  // One-time bf16 conversion of activations-in and all weights.
  cvt(x,    XBUF, (long long)T_ * B_ * I_);
  cvt(wih0, WIH0, (long long)G_ * I_);
  cvt(wihr, WIHR, (long long)(L_ - 1) * G_ * H_);
  cvt(whh,  WHH,  (long long)L_ * G_ * H_);

  for (int l = 0; l < L_; ++l) {
    const int   D    = (l == 0) ? I_ : H_;
    const bf16* Wl   = (l == 0) ? WIH0 : WIHR + (size_t)(l - 1) * G_ * H_;
    const int   last = (l == L_ - 1);

    // gi = layer_input @ w_ih.T + b_ih   (parallel over all T)
    gi_gemm<<<dim3((T_ * B_) / 64, G_ / 64), 256, 0, stream>>>(
        XBUF, D, Wl, bih + (size_t)l * G_, GI);

    // h <- hidden[l]
    init_h<<<(B_ * H_ + 255) / 256, 256, 0, stream>>>(
        hid + (size_t)l * B_ * H_, HST, HBF0, B_ * H_);

    const bf16* WHHl = WHH + (size_t)l * G_ * H_;
    for (int t = 0; t < T_; ++t) {
      const bf16* hin = (t & 1) ? HBF1 : HBF0;
      bf16* hout      = (t & 1) ? HBF0 : HBF1;
      gru_step<<<dim3(B_ / 32, H_ / 64), 256, 0, stream>>>(
          hin, hout, WHHl, GI + (size_t)t * B_ * G_, bhh + (size_t)l * G_,
          lens, HST,
          last ? out : nullptr,
          last ? nullptr : XBUF,  // layer output becomes next layer's input
          t, last);
    }

    // h_finals[l] = h
    (void)hipMemcpyAsync(out + (size_t)T_ * B_ * H_ + (size_t)l * B_ * H_, HST,
                         (size_t)B_ * H_ * 4, hipMemcpyDeviceToDevice, stream);
  }
}
